// Mypnufft_grasp_57853209477812
// MI455X (gfx1250) — matrix operand order repaired and checked by reading, not compiled, not run on hardware
//
#include <hip/hip_runtime.h>

typedef __attribute__((ext_vector_type(16))) _Float16 v16h;
typedef __attribute__((ext_vector_type(8)))  float    v8f;

#define NN      256         // image size N
#define NCOIL   8
#define MTOT    17408       // NSPOKE*NVEC = 34*512
#define KTILE   16          // k-points per workgroup
#define THREADS 256         // 8 waves of 32

// -------------------------------------------------------------------------
// Prep: src[c,a,b] = (Xr + iXi)*(Cr + iCi) -> f16 real/imag planes in ws.
// -------------------------------------------------------------------------
__global__ void prep_src_kernel(const float* __restrict__ X,
                                const float* __restrict__ C,
                                _Float16* __restrict__ srcR,
                                _Float16* __restrict__ srcI) {
    int idx = blockIdx.x * blockDim.x + threadIdx.x;   // over NCOIL*NN*NN
    if (idx >= NCOIL * NN * NN) return;
    int ab = idx % (NN * NN);
    float xr = X[2 * ab],  xi = X[2 * ab + 1];
    float cr = C[2 * idx], ci = C[2 * idx + 1];
    srcR[idx] = (_Float16)(xr * cr - xi * ci);
    srcI[idx] = (_Float16)(xr * ci + xi * cr);
}

// -------------------------------------------------------------------------
// Main: per (k-tile, coil): tmp = Eb x src^T via f16 WMMA (f32 accum),
// then y[k] = sum_a Ea[k,a]*tmp[k,a], scaled by w.
// -------------------------------------------------------------------------
__launch_bounds__(THREADS)
__global__ void nufft_wmma_kernel(const float* __restrict__ angles,
                                  const float* __restrict__ w,
                                  const _Float16* __restrict__ srcR,
                                  const _Float16* __restrict__ srcI,
                                  float* __restrict__ out) {
    // Eb tiles in A-fragment-friendly row-major [k][b] layout, f16.
    __shared__ __align__(16) _Float16 sEbR [KTILE * NN];  // cos(t*xb)
    __shared__ __align__(16) _Float16 sEbI [KTILE * NN];  // -sin(t*xb)
    __shared__ __align__(16) _Float16 sEbIn[KTILE * NN];  // +sin(t*xb)
    __shared__ float sYR[KTILE];
    __shared__ float sYI[KTILE];

    const int kt  = blockIdx.x;          // k tile index, 0..1087
    const int c   = blockIdx.y;          // coil
    const int k0  = kt * KTILE;
    const int tid = threadIdx.x;

    if (tid < KTILE) { sYR[tid] = 0.0f; sYI[tid] = 0.0f; }

    // ---- Phase A: build Eb = exp(-i * t_k * x_b) for this k tile ----
    for (int e = tid; e < KTILE * NN; e += THREADS) {
        int   kr = e >> 8;               // local k row (0..15)
        int   b  = e & (NN - 1);
        float t  = angles[(k0 + kr) * 2 + 1];
        float xb = (float)(b - NN / 2);
        float sn, cs;
        __sincosf(t * xb, &sn, &cs);
        sEbR [e] = (_Float16)cs;
        sEbI [e] = (_Float16)(-sn);
        sEbIn[e] = (_Float16)sn;
    }
    __syncthreads();

    const int wv   = tid >> 5;           // wave 0..7
    const int lane = tid & 31;
    const int row  = lane & 15;          // M index (k) for A, N index (a) for B
    const int half = lane >> 4;

    v8f accR[2], accI[2];
    for (int t = 0; t < 2; ++t)
        for (int r = 0; r < 8; ++r) { accR[t][r] = 0.0f; accI[t][r] = 0.0f; }

    const _Float16* baseR = srcR + (size_t)c * NN * NN;
    const _Float16* baseI = srcI + (size_t)c * NN * NN;

    // A-fragment LDS offset (halves): 16-bit 16x32 A layout ->
    // lane holds M=row; K runs [half*8, half*8+8) and [16+half*8, +8).
    const int offA = row * NN + half * 8;

    for (int j = 0; j < NN / 32; ++j) {              // K loop over b, step 32
        v16h aR, aI, aIn;
        {
            const _Float16* pA = sEbR + offA + j * 32;
            ((float4*)&aR)[0] = *(const float4*)(pA);
            ((float4*)&aR)[1] = *(const float4*)(pA + 16);
            const _Float16* pB = sEbI + offA + j * 32;
            ((float4*)&aI)[0] = *(const float4*)(pB);
            ((float4*)&aI)[1] = *(const float4*)(pB + 16);
            const _Float16* pC = sEbIn + offA + j * 32;
            ((float4*)&aIn)[0] = *(const float4*)(pC);
            ((float4*)&aIn)[1] = *(const float4*)(pC + 16);
        }

#pragma unroll
        for (int t = 0; t < 2; ++t) {
            const int aIdx = (wv * 2 + t) * 16 + row;        // B: N = a
            // B-fragment: lane holds N=row, K runs [half*8, half*8+8) and [16+half*8, +8).
            const _Float16* pR = baseR + (size_t)aIdx * NN + j * 32 + half * 8;
            const _Float16* pI = baseI + (size_t)aIdx * NN + j * 32 + half * 8;
            v16h bR, bI;
            ((float4*)&bR)[0] = *(const float4*)(pR);
            ((float4*)&bR)[1] = *(const float4*)(pR + 16);
            ((float4*)&bI)[0] = *(const float4*)(pI);
            ((float4*)&bI)[1] = *(const float4*)(pI + 16);

            if (j + 1 < NN / 32) {                  // hint next K step
                __builtin_prefetch(pR + 32, 0, 1);
                __builtin_prefetch(pI + 32, 0, 1);
            }

            // Complex MAC (f16 WMMA forbids A/B NEG -> use +sin copy):
            // tmpR += EbR*SR + sin*SI ;  tmpI += EbR*SI + EbI*SR
            accR[t] = __builtin_amdgcn_wmma_f32_16x16x32_f16(
                false, aR,  false, bR, (short)0, accR[t], false, false);
            accR[t] = __builtin_amdgcn_wmma_f32_16x16x32_f16(
                false, aIn, false, bI, (short)0, accR[t], false, false);
            accI[t] = __builtin_amdgcn_wmma_f32_16x16x32_f16(
                false, aR,  false, bI, (short)0, accI[t], false, false);
            accI[t] = __builtin_amdgcn_wmma_f32_16x16x32_f16(
                false, aI,  false, bR, (short)0, accI[t], false, false);
        }
    }

    // ---- Epilogue: y[k] = sum_a Ea[k,a] * tmp[k,a] ----
    // C/D layout: VGPR r, lane -> k = r + 8*half, a = tile*16 + row.
#pragma unroll
    for (int t = 0; t < 2; ++t) {
        const int   aIdx = (wv * 2 + t) * 16 + row;
        const float xa   = (float)(aIdx - NN / 2);
#pragma unroll
        for (int r = 0; r < 8; ++r) {
            int   kk = r + 8 * half;                 // local k in tile
            float s  = angles[(k0 + kk) * 2];
            float sn, cs;
            __sincosf(s * xa, &sn, &cs);
            float eaR = cs, eaI = -sn;
            float tR = accR[t][r], tI = accI[t][r];
            atomicAdd(&sYR[kk], eaR * tR - eaI * tI);   // ds_add_f32
            atomicAdd(&sYI[kk], eaR * tI + eaI * tR);
        }
    }
    __syncthreads();

    if (tid < KTILE) {
        int    k  = k0 + tid;
        float  wk = w[k];
        size_t o  = ((size_t)c * MTOT + k) * 2;
        out[o + 0] = sYR[tid] * wk;
        out[o + 1] = sYI[tid] * wk;
    }
}

// -------------------------------------------------------------------------
extern "C" void kernel_launch(void* const* d_in, const int* in_sizes, int n_in,
                              void* d_out, int out_size, void* d_ws, size_t ws_size,
                              hipStream_t stream) {
    (void)in_sizes; (void)n_in; (void)out_size; (void)ws_size;
    const float* X      = (const float*)d_in[0];   // (256,256,2)
    const float* angles = (const float*)d_in[1];   // (M,2)
    const float* C      = (const float*)d_in[2];   // (8,256,256,2)
    const float* w      = (const float*)d_in[3];   // (34,512)
    // d_in[4..6] = Nspoke, Nvec, Nc scalars (compile-time constants here)

    _Float16* srcR = (_Float16*)d_ws;                       // 1 MB
    _Float16* srcI = srcR + (size_t)NCOIL * NN * NN;        // 1 MB

    dim3 gPrep((NCOIL * NN * NN + THREADS - 1) / THREADS);
    prep_src_kernel<<<gPrep, THREADS, 0, stream>>>(X, C, srcR, srcI);

    dim3 gMain(MTOT / KTILE, NCOIL);
    nufft_wmma_kernel<<<gMain, THREADS, 0, stream>>>(angles, w, srcR, srcI,
                                                     (float*)d_out);
}
